// MultiHeadAttention_65060164600355
// MI455X (gfx1250) — compile-verified, matrix operand order checked by reading, not run on hardware
//
#include <hip/hip_runtime.h>

// ---------------------------------------------------------------------------
// MultiHeadAttention on gfx1250 (MI455X): bf16 WMMA + async-LDS pipelines.
// B=4, S=2048, D=1024, H=16, DH=64
// ---------------------------------------------------------------------------

#define B_  4
#define S_  2048
#define D_  1024
#define H_  16
#define DH_ 64
#define MROWS (B_ * S_)   // 8192

typedef __attribute__((ext_vector_type(16))) __bf16 v16bf;
typedef __attribute__((ext_vector_type(8)))  __bf16 v8bf;
typedef __attribute__((ext_vector_type(4)))  __bf16 v4bf;
typedef __attribute__((ext_vector_type(8)))  float  v8f;

__device__ __forceinline__ v8f wmma_bf16(v16bf a, v16bf b, v8f c) {
    return __builtin_amdgcn_wmma_f32_16x16x32_bf16(
        false, a, false, b, (short)0, c, false, false);
}

__device__ __forceinline__ v8f v8f_zero() {
    v8f z;
#pragma unroll
    for (int e = 0; e < 8; ++e) z[e] = 0.0f;
    return z;
}

// ---- CDNA5 async global->LDS (per-lane 16B, tracked with ASYNCcnt) --------
__device__ __forceinline__ void async_cp16(const __bf16* lds_ptr, const __bf16* gptr) {
    unsigned int lds_addr = (unsigned int)(unsigned long long)lds_ptr; // addr[31:0] = LDS offset
    asm volatile("global_load_async_to_lds_b128 %0, %1, off"
                 :: "v"(lds_addr), "v"(gptr)
                 : "memory");
}
__device__ __forceinline__ void wait_async0() { asm volatile("s_wait_asynccnt 0x0" ::: "memory"); }
__device__ __forceinline__ void wait_async1() { asm volatile("s_wait_asynccnt 0x1" ::: "memory"); }
__device__ __forceinline__ void wait_async4() { asm volatile("s_wait_asynccnt 0x4" ::: "memory"); }

// Load a 16x32 bf16 tile (row-major, stride ld elements) into the WMMA
// 16-bit A-operand layout (also valid for B when source is B-transposed):
//   lane L: row = L&15, half h = L>>4
//   elems 0..7  = cols 8h..8h+7 ; elems 8..15 = cols 16+8h..16+8h+7
__device__ __forceinline__ v16bf load_tile_bf16(const __bf16* __restrict__ base, int ld) {
    const int lane = (int)(threadIdx.x & 31u);
    const int r = lane & 15;
    const int h = lane >> 4;
    const __bf16* p = base + (size_t)r * (size_t)ld + 8 * h;
    v8bf c0 = *(const v8bf*)(p);
    v8bf c1 = *(const v8bf*)(p + 16);
    v16bf out;
#pragma unroll
    for (int e = 0; e < 8; ++e) { out[e] = c0[e]; out[8 + e] = c1[e]; }
    return out;
}

// --------------------------- pre-pass kernels ------------------------------

__global__ __launch_bounds__(256) void cvt_f32_to_bf16(const float* __restrict__ in,
                                                       __bf16* __restrict__ out, int n) {
    int i = ((int)blockIdx.x * (int)blockDim.x + (int)threadIdx.x) * 4;
    if (i + 3 < n) {
        float4 v = *(const float4*)(in + i);
        v4bf o;
        o[0] = (__bf16)v.x; o[1] = (__bf16)v.y; o[2] = (__bf16)v.z; o[3] = (__bf16)v.w;
        *(v4bf*)(out + i) = o;
    }
}

// W[k][n] fp32 -> Wt[n][k] bf16
__global__ __launch_bounds__(256) void transpose_w_bf16(const float* __restrict__ W,
                                                        __bf16* __restrict__ Wt) {
    int idx = (int)blockIdx.x * (int)blockDim.x + (int)threadIdx.x;
    int k = idx >> 10;
    int n = idx & 1023;
    Wt[(size_t)n * D_ + k] = (__bf16)W[(size_t)k * D_ + n];
}

// --------------------------- GEMM core -------------------------------------
// Block: 256 threads = 8 waves, computes 128x64 tile of Y = X @ W.
// B tile (Bt rows n0..n0+63, 32 k) staged into LDS via async DMA, double
// buffered; A-operand register double buffer overlaps global loads with WMMA.
template <int MODE>  // 0: q-headmajor*scale, 1: k-headmajor, 2: v-transposed, 3: f32 rowmajor out
__global__ __launch_bounds__(256) void gemm_kernel(const __bf16* __restrict__ X,
                                                   const __bf16* __restrict__ Wt,
                                                   __bf16* __restrict__ dstb,
                                                   float* __restrict__ dstf,
                                                   float scale) {
    __shared__ __bf16 Bl[2][64 * 32];   // [n_local][k_local], 4 KB each

    const int t   = (int)threadIdx.x;
    const int wid = t >> 5;
    const int bm  = (int)blockIdx.x >> 4;     // 0..63
    const int bn  = (int)blockIdx.x & 15;     // 0..15
    const int m0  = bm * 128 + wid * 16;
    const int n0  = bn * 64;

    // staging map: thread t -> one 16B chunk of the 64x32 B tile
    const int srow = t >> 2;          // 0..63
    const int sc   = (t & 3) * 8;     // col offset 0,8,16,24

    // stage k-step 0
    async_cp16(&Bl[0][srow * 32 + sc], Wt + (size_t)(n0 + srow) * D_ + sc);

    v8f acc[4];
#pragma unroll
    for (int j = 0; j < 4; ++j) acc[j] = v8f_zero();

    v16bf a_cur = load_tile_bf16(X + (size_t)m0 * D_, D_);

    for (int kk = 0; kk < D_ / 32; ++kk) {
        const int cur = kk & 1;
        if (kk + 1 < D_ / 32) {
            async_cp16(&Bl[cur ^ 1][srow * 32 + sc],
                       Wt + (size_t)(n0 + srow) * D_ + (kk + 1) * 32 + sc);
            wait_async1();
        } else {
            wait_async0();
        }
        __syncthreads();                       // Bl[cur] ready for all waves

        v16bf b0 = load_tile_bf16(&Bl[cur][0 * 16 * 32], 32);
        v16bf b1 = load_tile_bf16(&Bl[cur][1 * 16 * 32], 32);
        v16bf b2 = load_tile_bf16(&Bl[cur][2 * 16 * 32], 32);
        v16bf b3 = load_tile_bf16(&Bl[cur][3 * 16 * 32], 32);

        v16bf a_nxt = a_cur;
        if (kk + 1 < D_ / 32)
            a_nxt = load_tile_bf16(X + (size_t)m0 * D_ + (kk + 1) * 32, D_);

        acc[0] = wmma_bf16(a_cur, b0, acc[0]);
        acc[1] = wmma_bf16(a_cur, b1, acc[1]);
        acc[2] = wmma_bf16(a_cur, b2, acc[2]);
        acc[3] = wmma_bf16(a_cur, b3, acc[3]);
        a_cur = a_nxt;

        __syncthreads();                       // all waves done with Bl[cur]
    }

    const int lane = t & 31;
    const int nl   = lane & 15;
    const int half = lane >> 4;
#pragma unroll
    for (int j = 0; j < 4; ++j) {
#pragma unroll
        for (int i = 0; i < 8; ++i) {
            int m = m0 + 8 * half + i;
            int n = n0 + 16 * j + nl;
            float v = acc[j][i] * scale;
            if (MODE == 3) {
                dstf[(size_t)m * D_ + n] = v;
            } else {
                int b = m >> 11, s = m & (S_ - 1);
                int h = n >> 6,  d = n & (DH_ - 1);
                if (MODE == 2) dstb[(((size_t)b * H_ + h) * DH_ + d) * S_ + s] = (__bf16)v;
                else           dstb[(((size_t)b * H_ + h) * S_ + s) * DH_ + d] = (__bf16)v;
            }
        }
    }
}

// --------------------------- flash attention -------------------------------
// Block: 128 threads = 4 waves, all on the same (b,h); waves own consecutive
// 16-row q tiles. K/V tiles for each 32-key block are async-staged into LDS
// once per block (shared by the 4 waves), double buffered.
__global__ __launch_bounds__(128) void flash_attn(const __bf16* __restrict__ Qh,
                                                  const __bf16* __restrict__ Kh,
                                                  const __bf16* __restrict__ Vt,
                                                  __bf16* __restrict__ ctx) {
    __shared__ __bf16 Kl[2][32 * 64];    // [key][dim]  4 KB each
    __shared__ __bf16 Vl[2][64 * 32];    // [dim][key]  4 KB each
    __shared__ __bf16 Plds[4][16 * 32];  // per-wave P transpose tile

    const int t   = (int)threadIdx.x;
    const int wid = t >> 5;
    const int wg  = (int)blockIdx.x * 4 + wid;   // 0..8191
    const int qt  = wg & (S_ / 16 - 1);
    const int bh  = wg >> 7;
    const int b   = bh >> 4, h = bh & 15;

    const __bf16* Qb = Qh + (size_t)bh * S_ * DH_ + (size_t)qt * 16 * DH_;
    const __bf16* Kb = Kh + (size_t)bh * S_ * DH_;
    const __bf16* Vb = Vt + (size_t)bh * DH_ * S_;

    const int lane = t & 31;
    const int nl   = lane & 15;
    const int half = lane >> 4;

    // staging maps (128 threads, 2 chunks each per tile)
    const int krow0 = t >> 3,          kc0 = (t & 7) * 8;          // K chunks 0..127
    const int krow1 = (t + 128) >> 3,  kc1 = kc0;                  // K chunks 128..255
    const int vrow0 = t >> 2,          vc0 = (t & 3) * 8;          // V chunks 0..127
    const int vrow1 = (t + 128) >> 2,  vc1 = vc0;                  // V chunks 128..255

    // stage key-block 0 into buffer 0 (4 async instructions / thread)
    {
        async_cp16(&Kl[0][krow0 * 64 + kc0], Kb + (size_t)krow0 * DH_ + kc0);
        async_cp16(&Kl[0][krow1 * 64 + kc1], Kb + (size_t)krow1 * DH_ + kc1);
        async_cp16(&Vl[0][vrow0 * 32 + vc0], Vb + (size_t)vrow0 * S_ + vc0);
        async_cp16(&Vl[0][vrow1 * 32 + vc1], Vb + (size_t)vrow1 * S_ + vc1);
    }

    v16bf qa0 = load_tile_bf16(Qb, DH_);
    v16bf qa1 = load_tile_bf16(Qb + 32, DH_);

    v8f o0 = v8f_zero(), o1 = v8f_zero(), o2 = v8f_zero(), o3 = v8f_zero();
    float mrow[8], lrow[8];
#pragma unroll
    for (int i = 0; i < 8; ++i) { mrow[i] = -1e30f; lrow[i] = 0.0f; }

    __bf16* Pw = &Plds[wid][0];

    for (int kb = 0; kb < S_ / 32; ++kb) {
        const int cur = kb & 1;
        if (kb + 1 < S_ / 32) {
            const int k0n = (kb + 1) * 32;
            async_cp16(&Kl[cur ^ 1][krow0 * 64 + kc0], Kb + (size_t)(k0n + krow0) * DH_ + kc0);
            async_cp16(&Kl[cur ^ 1][krow1 * 64 + kc1], Kb + (size_t)(k0n + krow1) * DH_ + kc1);
            async_cp16(&Vl[cur ^ 1][vrow0 * 32 + vc0], Vb + (size_t)vrow0 * S_ + k0n + vc0);
            async_cp16(&Vl[cur ^ 1][vrow1 * 32 + vc1], Vb + (size_t)vrow1 * S_ + k0n + vc1);
            wait_async4();
        } else {
            wait_async0();
        }
        __syncthreads();                     // Kl/Vl[cur] ready

        // scores for the 32 keys (two 16x16 C tiles)
        v8f s0 = v8f_zero(), s1 = v8f_zero();
        {
            v16bf b0 = load_tile_bf16(&Kl[cur][0],            DH_);
            v16bf b1 = load_tile_bf16(&Kl[cur][32],           DH_);
            s0 = wmma_bf16(qa0, b0, s0);
            s0 = wmma_bf16(qa1, b1, s0);
            v16bf b2 = load_tile_bf16(&Kl[cur][16 * 64],      DH_);
            v16bf b3 = load_tile_bf16(&Kl[cur][16 * 64 + 32], DH_);
            s1 = wmma_bf16(qa0, b2, s1);
            s1 = wmma_bf16(qa1, b3, s1);
        }

        // online softmax across the 32 new columns (row i lives in elem i,
        // spread over the 16 lanes of this half -> xor-shuffle reductions)
#pragma unroll
        for (int i = 0; i < 8; ++i) {
            float mx = fmaxf(s0[i], s1[i]);
            mx = fmaxf(mx, __shfl_xor(mx, 1, 32));
            mx = fmaxf(mx, __shfl_xor(mx, 2, 32));
            mx = fmaxf(mx, __shfl_xor(mx, 4, 32));
            mx = fmaxf(mx, __shfl_xor(mx, 8, 32));
            float mn    = fmaxf(mrow[i], mx);
            float alpha = __expf(mrow[i] - mn);
            mrow[i]     = mn;
            float p0 = __expf(s0[i] - mn);
            float p1 = __expf(s1[i] - mn);
            float rs = p0 + p1;
            rs += __shfl_xor(rs, 1, 32);
            rs += __shfl_xor(rs, 2, 32);
            rs += __shfl_xor(rs, 4, 32);
            rs += __shfl_xor(rs, 8, 32);
            lrow[i] = lrow[i] * alpha + rs;
            o0[i] *= alpha; o1[i] *= alpha; o2[i] *= alpha; o3[i] *= alpha;
            Pw[(8 * half + i) * 32 + nl]      = (__bf16)p0;
            Pw[(8 * half + i) * 32 + 16 + nl] = (__bf16)p1;
        }

        // re-read P in A-operand layout (per-wave LDS; DS ops in-order)
        v16bf pa;
        {
            const __bf16* pr = Pw + nl * 32 + 8 * half;
#pragma unroll
            for (int e = 0; e < 8; ++e) { pa[e] = pr[e]; pa[8 + e] = pr[16 + e]; }
        }

        // o += P[16x32] @ V[32x64]
        o0 = wmma_bf16(pa, load_tile_bf16(&Vl[cur][0 * 16 * 32], 32), o0);
        o1 = wmma_bf16(pa, load_tile_bf16(&Vl[cur][1 * 16 * 32], 32), o1);
        o2 = wmma_bf16(pa, load_tile_bf16(&Vl[cur][2 * 16 * 32], 32), o2);
        o3 = wmma_bf16(pa, load_tile_bf16(&Vl[cur][3 * 16 * 32], 32), o3);

        __syncthreads();                     // all waves done with Kl/Vl[cur]
    }

    // normalize + store merged-head ctx [B,S,D] bf16
#pragma unroll
    for (int i = 0; i < 8; ++i) {
        float inv = 1.0f / lrow[i];
        int m = 8 * half + i;
        size_t ro = ((size_t)b * S_ + (size_t)qt * 16 + m) * D_ + (size_t)h * DH_ + nl;
        ctx[ro + 0]  = (__bf16)(o0[i] * inv);
        ctx[ro + 16] = (__bf16)(o1[i] * inv);
        ctx[ro + 32] = (__bf16)(o2[i] * inv);
        ctx[ro + 48] = (__bf16)(o3[i] * inv);
    }
}

// --------------------------- host launcher ---------------------------------

extern "C" void kernel_launch(void* const* d_in, const int* in_sizes, int n_in,
                              void* d_out, int out_size, void* d_ws, size_t ws_size,
                              hipStream_t stream) {
    (void)in_sizes; (void)n_in; (void)out_size; (void)ws_size;

    const float* q_in = (const float*)d_in[0];
    const float* k_in = (const float*)d_in[1];
    const float* v_in = (const float*)d_in[2];
    const float* Wq   = (const float*)d_in[3];
    const float* Wk   = (const float*)d_in[4];
    const float* Wv   = (const float*)d_in[5];
    const float* Wo   = (const float*)d_in[6];

    char* ws = (char*)d_ws;
    const size_t SZ_ACT = (size_t)MROWS * D_ * 2;   // 16 MB bf16
    const size_t SZ_W   = (size_t)D_ * D_ * 2;      //  2 MB bf16

    __bf16* xq  = (__bf16*)(ws);
    __bf16* xk  = (__bf16*)(ws + SZ_ACT);
    __bf16* xv  = (__bf16*)(ws + 2 * SZ_ACT);
    __bf16* Wqt = (__bf16*)(ws + 3 * SZ_ACT);
    __bf16* Wkt = (__bf16*)(ws + 3 * SZ_ACT + SZ_W);
    __bf16* Wvt = (__bf16*)(ws + 3 * SZ_ACT + 2 * SZ_W);
    __bf16* Wot = (__bf16*)(ws + 3 * SZ_ACT + 3 * SZ_W);
    __bf16* Qh  = (__bf16*)(ws + 3 * SZ_ACT + 4 * SZ_W);
    __bf16* Kh  = (__bf16*)(ws + 4 * SZ_ACT + 4 * SZ_W);
    __bf16* Vt  = (__bf16*)(ws + 5 * SZ_ACT + 4 * SZ_W);
    __bf16* ctx = xq;  // xq fully consumed before flash_attn writes ctx

    const int nAct = MROWS * D_;
    cvt_f32_to_bf16<<<nAct / 1024, 256, 0, stream>>>(q_in, xq, nAct);
    cvt_f32_to_bf16<<<nAct / 1024, 256, 0, stream>>>(k_in, xk, nAct);
    cvt_f32_to_bf16<<<nAct / 1024, 256, 0, stream>>>(v_in, xv, nAct);

    const int wBlocks = (D_ * D_) / 256;
    transpose_w_bf16<<<wBlocks, 256, 0, stream>>>(Wq, Wqt);
    transpose_w_bf16<<<wBlocks, 256, 0, stream>>>(Wk, Wkt);
    transpose_w_bf16<<<wBlocks, 256, 0, stream>>>(Wv, Wvt);
    transpose_w_bf16<<<wBlocks, 256, 0, stream>>>(Wo, Wot);

    // GEMM grid: (8192/128) * (1024/64) = 1024 blocks, 256 threads
    const float scale = 0.03125f;  // 1/sqrt(1024)
    gemm_kernel<0><<<1024, 256, 0, stream>>>(xq, Wqt, Qh, nullptr, scale);
    gemm_kernel<1><<<1024, 256, 0, stream>>>(xk, Wkt, Kh, nullptr, 1.0f);
    gemm_kernel<2><<<1024, 256, 0, stream>>>(xv, Wvt, Vt, nullptr, 1.0f);

    // flash grid: B*H*(S/16)/4 = 2048 blocks, 128 threads
    flash_attn<<<2048, 128, 0, stream>>>(Qh, Kh, Vt, ctx);

    gemm_kernel<3><<<1024, 256, 0, stream>>>(ctx, Wot, nullptr, (float*)d_out, 1.0f);
}
